// GPT2Attention_45475113730359
// MI455X (gfx1250) — matrix operand order repaired and checked by reading, not compiled, not run on hardware
//
#include <hip/hip_runtime.h>
#include <hip/hip_bf16.h>

// ---------------------------------------------------------------------------
// GPT-2 attention block for MI455X (gfx1250, wave32, WMMA, async-to-LDS).
//   1) transpose_cvt: f32 weights -> f16, [K][N] -> [N][K] (B-frag friendly)
//   2) qkv_gemm:  hidden x Wqkv -> Q[b,h,s,d], K[b,h,s,d], V[b,h,d,s] (f16,
//                 bias fused). B tiles staged in LDS via
//                 global_load_async_to_lds_b128 (ASYNCcnt), double-buffered.
//   3) attn:      flash-style causal attention, WMMA for QK^T and PV,
//                 online softmax in registers, P re-layout via LDS bounce.
//   4) proj_gemm: attn_out x Wproj + bias -> f32, same async-staged B tiles.
// ---------------------------------------------------------------------------

typedef __attribute__((ext_vector_type(16))) _Float16 v16h;
typedef __attribute__((ext_vector_type(8)))  _Float16 h8;
typedef __attribute__((ext_vector_type(8)))  float    v8f;
typedef __attribute__((ext_vector_type(4)))  float    f32x4;

#define WMMA16(A, B, C) \
  __builtin_amdgcn_wmma_f32_16x16x32_f16(false, (A), false, (B), (short)0, (C), false, false)

static __device__ __forceinline__ v16h pack_afrag(h8 a, h8 b) {
  // A-fragment (16x32 f16): elements 0..7 = first K-octet, 8..15 = second.
  v16h r;
#pragma unroll
  for (int i = 0; i < 8; ++i) { r[i] = a[i]; r[i + 8] = b[i]; }
  return r;
}

// Async copy of 16 bytes global -> LDS, tracked by ASYNCcnt (CDNA5 TDM-lite
// path).  lds_off = wave-relative LDS byte address (flat addr low 32 bits).
static __device__ __forceinline__ void async_g2l_b128(unsigned lds_off,
                                                      const void* gptr) {
  asm volatile("global_load_async_to_lds_b128 %0, %1, off"
               :
               : "v"(lds_off), "v"((unsigned long long)gptr)
               : "memory");
}
static __device__ __forceinline__ unsigned lds_addr32(const void* p) {
  return (unsigned)(unsigned long long)p;  // LDS aperture: offset = addr[31:0]
}

// --------------------------- weight transpose ------------------------------
// dst[n*K + k] = (f16) src[k*N + n];  K,N multiples of 32.
__global__ __launch_bounds__(256) void transpose_cvt(const float* __restrict__ src,
                                                     _Float16* __restrict__ dst,
                                                     int K, int N) {
  __shared__ _Float16 tile[32][33];
  const int tx = threadIdx.x & 31;
  const int ty = threadIdx.x >> 5;      // 0..7
  const int nb = blockIdx.x * 32;
  const int kb = blockIdx.y * 32;
#pragma unroll
  for (int i = 0; i < 32; i += 8) {
    tile[ty + i][tx] = (_Float16)src[(size_t)(kb + ty + i) * N + (nb + tx)];
  }
  __syncthreads();
#pragma unroll
  for (int i = 0; i < 32; i += 8) {
    dst[(size_t)(nb + ty + i) * K + (kb + tx)] = tile[tx][ty + i];
  }
}

// ------------------------------ QKV GEMM -----------------------------------
// A: hidden f32 [4096][1024], Bt: f16 [3072][1024], bias f32 [3072].
// Block = 128 thr (4 waves); block tile 64(M) x 64(N); wave = 16(M) x 64(N).
// B tile (64x32 f16 = 4KB) async-staged into LDS, double buffered.
__global__ __launch_bounds__(128) void qkv_gemm(const float* __restrict__ A,
                                                const _Float16* __restrict__ Bt,
                                                const float* __restrict__ bias,
                                                _Float16* __restrict__ qws,
                                                _Float16* __restrict__ kws,
                                                _Float16* __restrict__ vws) {
  __shared__ __align__(16) _Float16 bstage[2][64 * 32];
  const int tid = threadIdx.x;
  const int lane = tid & 31;
  const int wave = tid >> 5;
  const int lo = lane & 15, hi = lane >> 4;
  const int M0 = blockIdx.x * 64 + wave * 16;
  const int N0 = blockIdx.y * 64;
  const int K = 1024;

  // Async-stage mapping: thread -> 32B unit (row n = tid/2, half = tid&1).
  const int srow = tid >> 1, shalf = tid & 1;
  const _Float16* bsrc = Bt + (size_t)(N0 + srow) * K + shalf * 16;
  const unsigned bdst0 = lds_addr32(&bstage[0][srow * 32 + shalf * 16]);
  const unsigned bdst1 = lds_addr32(&bstage[1][srow * 32 + shalf * 16]);

  const float* arow = A + (size_t)(M0 + lo) * K;
  v8f acc[4] = {};

  // Prologue: stage k0 = 0 into buffer 0.
  async_g2l_b128(bdst0, bsrc);
  async_g2l_b128(bdst0 + 16, (const char*)bsrc + 16);

  for (int k0 = 0; k0 < K; k0 += 32) {
    const int cur = (k0 >> 5) & 1;
    if (k0 + 32 < K) {  // stage next tile while current one is consumed
      const unsigned nd = cur ? bdst0 : bdst1;
      const _Float16* ns = bsrc + (k0 + 32);
      async_g2l_b128(nd, ns);
      async_g2l_b128(nd + 16, (const char*)ns + 16);
      asm volatile("s_wait_asynccnt 2" ::: "memory");  // current tile landed
    } else {
      asm volatile("s_wait_asynccnt 0" ::: "memory");
    }
    __syncthreads();  // all waves' portions of current tile visible

    if (k0 + 64 < K) __builtin_prefetch(arow + k0 + 64, 0, 1);

    // A-fragment: f32 -> f16 on the fly (per-wave rows, global direct).
    const int d0 = k0 + 16 * hi;
    f32x4 a0 = *(const f32x4*)(arow + d0);
    f32x4 a1 = *(const f32x4*)(arow + d0 + 4);
    f32x4 a2 = *(const f32x4*)(arow + d0 + 8);
    f32x4 a3 = *(const f32x4*)(arow + d0 + 12);
    v16h af;
#pragma unroll
    for (int i = 0; i < 4; ++i) {
      af[i]      = (_Float16)a0[i];
      af[i + 4]  = (_Float16)a1[i];
      af[i + 8]  = (_Float16)a2[i];
      af[i + 12] = (_Float16)a3[i];
    }
#pragma unroll
    for (int t = 0; t < 4; ++t) {
      const v16h bf = *(const v16h*)(&bstage[cur][(16 * t + lo) * 32 + 16 * hi]);
      acc[t] = WMMA16(af, bf, acc[t]);
    }
    __syncthreads();  // done consuming before next overwrite
  }

  // Epilogue: bias, f16 convert, scatter into head-major Q/K/V (V transposed).
#pragma unroll
  for (int t = 0; t < 4; ++t) {
    const int n = N0 + 16 * t + lo;           // 0..3071
    const float bv = bias[n];
    const int seg = n >> 10;                  // 0=Q 1=K 2=V
    const int nn = n & 1023;
    const int h = nn >> 6, d = nn & 63;
#pragma unroll
    for (int r = 0; r < 8; ++r) {
      const int row = M0 + r + 8 * hi;        // b*2048 + s
      const int b = row >> 11, s = row & 2047;
      const _Float16 val = (_Float16)(acc[t][r] + bv);
      const size_t bh = (size_t)(b * 16 + h);
      if (seg == 0)      qws[(bh * 2048 + s) * 64 + d] = val;
      else if (seg == 1) kws[(bh * 2048 + s) * 64 + d] = val;
      else               vws[(bh * 64 + d) * 2048 + s] = val;
    }
  }
}

// --------------------------- flash attention -------------------------------
// One wave per 16-query tile; causal online-softmax; 8 WMMA per 32-key chunk.
__global__ __launch_bounds__(128) void attn_kernel(const _Float16* __restrict__ qws,
                                                   const _Float16* __restrict__ kws,
                                                   const _Float16* __restrict__ vws,
                                                   _Float16* __restrict__ ows) {
  __shared__ __align__(16) _Float16 pbuf[4][16 * 32];  // per-wave P bounce
  const int lane = threadIdx.x & 31;
  const int wave = threadIdx.x >> 5;
  const int lo = lane & 15, hi = lane >> 4;

  const int tid = blockIdx.x * 4 + wave;     // 0..4095 query tiles
  const int qt = tid & 127;
  const int h = (tid >> 7) & 15;
  const int b = tid >> 11;
  const int bh = b * 16 + h;
  const int qbase = qt * 16;

  const _Float16* Q = qws + (size_t)bh * 2048 * 64;
  const _Float16* Kp = kws + (size_t)bh * 2048 * 64;
  const _Float16* Vp = vws + (size_t)bh * 64 * 2048;
  _Float16* pb = &pbuf[wave][0];

  // Q fragments for both K-dim steps (HD=64 -> 2 x 32), loaded once.
  const _Float16* qrow = Q + (size_t)(qbase + lo) * 64;
  v16h qf[2];
#pragma unroll
  for (int s = 0; s < 2; ++s) {
    const int base = 32 * s + 16 * hi;
    const h8 a = *(const h8*)(qrow + base);
    const h8 c = *(const h8*)(qrow + base + 8);
    qf[s] = pack_afrag(a, c);
  }

  v8f o[4] = {};
  float mrow[8], lrow[8];
#pragma unroll
  for (int r = 0; r < 8; ++r) { mrow[r] = -3.0e38f; lrow[r] = 0.0f; }
  const float scale = 0.125f;  // 1/sqrt(64)

  for (int kb = 0; kb <= qbase; kb += 32) {   // causal: keys kb..kb+31
    // ---- scores: S = Q K^T (two 16x16 n-tiles, K-dim 64) ----
    v8f sc[2] = {};
#pragma unroll
    for (int t = 0; t < 2; ++t) {
      const _Float16* krow = Kp + (size_t)(kb + 16 * t + lo) * 64;
#pragma unroll
      for (int s = 0; s < 2; ++s) {
        const v16h bf = *(const v16h*)(krow + 32 * s + 16 * hi);
        sc[t] = WMMA16(qf[s], bf, sc[t]);
      }
    }

    // ---- online softmax (row stats per the C-matrix layout) ----
#pragma unroll
    for (int r = 0; r < 8; ++r) {
      const int qr = qbase + r + 8 * hi;
      float s0 = (kb + lo <= qr)      ? sc[0][r] * scale : -3.0e38f;
      float s1 = (kb + 16 + lo <= qr) ? sc[1][r] * scale : -3.0e38f;
      float cm = fmaxf(s0, s1);
      cm = fmaxf(cm, __shfl_xor(cm, 1, 16));
      cm = fmaxf(cm, __shfl_xor(cm, 2, 16));
      cm = fmaxf(cm, __shfl_xor(cm, 4, 16));
      cm = fmaxf(cm, __shfl_xor(cm, 8, 16));
      const float nm = fmaxf(mrow[r], cm);
      const float p0 = __expf(s0 - nm);
      const float p1 = __expf(s1 - nm);
      float rs = p0 + p1;
      rs += __shfl_xor(rs, 1, 16);
      rs += __shfl_xor(rs, 2, 16);
      rs += __shfl_xor(rs, 4, 16);
      rs += __shfl_xor(rs, 8, 16);
      const float corr = __expf(mrow[r] - nm);
      lrow[r] = lrow[r] * corr + rs;
      mrow[r] = nm;
#pragma unroll
      for (int t = 0; t < 4; ++t) o[t][r] *= corr;
      // Stash P row-major [row][col32] for A-fragment reload.
      pb[(r + 8 * hi) * 32 + lo] = (_Float16)p0;
      pb[(r + 8 * hi) * 32 + 16 + lo] = (_Float16)p1;
    }

    // ---- C-layout -> A-fragment via LDS bounce ----
    const h8 pa = *(const h8*)(pb + lo * 32 + 16 * hi);
    const h8 pc = *(const h8*)(pb + lo * 32 + 8 + 16 * hi);
    const v16h pf = pack_afrag(pa, pc);

    // ---- O += P V  (V stored [d][s] -> contiguous B-fragments) ----
#pragma unroll
    for (int t = 0; t < 4; ++t) {
      const v16h bf = *(const v16h*)(Vp + (size_t)(16 * t + lo) * 2048 + kb + 16 * hi);
      o[t] = WMMA16(pf, bf, o[t]);
    }
  }

  // ---- normalize and store merged-head f16 output [b,s,h*64+d] ----
  float inv[8];
#pragma unroll
  for (int r = 0; r < 8; ++r) inv[r] = 1.0f / lrow[r];
#pragma unroll
  for (int t = 0; t < 4; ++t) {
#pragma unroll
    for (int r = 0; r < 8; ++r) {
      const int row = qbase + r + 8 * hi;
      ows[((size_t)(b * 2048 + row)) * 1024 + h * 64 + 16 * t + lo] =
          (_Float16)(o[t][r] * inv[r]);
    }
  }
}

// ------------------------------ Proj GEMM ----------------------------------
// A: f16 [4096][1024] (attn out), Bt: f16 [1024][1024], out f32 + bias.
// Same async double-buffered B staging as qkv_gemm.
__global__ __launch_bounds__(128) void proj_gemm(const _Float16* __restrict__ A,
                                                 const _Float16* __restrict__ Bt,
                                                 const float* __restrict__ bias,
                                                 float* __restrict__ out) {
  __shared__ __align__(16) _Float16 bstage[2][64 * 32];
  const int tid = threadIdx.x;
  const int lane = tid & 31;
  const int wave = tid >> 5;
  const int lo = lane & 15, hi = lane >> 4;
  const int M0 = blockIdx.x * 64 + wave * 16;
  const int N0 = blockIdx.y * 64;
  const int K = 1024;

  const int srow = tid >> 1, shalf = tid & 1;
  const _Float16* bsrc = Bt + (size_t)(N0 + srow) * K + shalf * 16;
  const unsigned bdst0 = lds_addr32(&bstage[0][srow * 32 + shalf * 16]);
  const unsigned bdst1 = lds_addr32(&bstage[1][srow * 32 + shalf * 16]);

  const _Float16* arow = A + (size_t)(M0 + lo) * K;
  v8f acc[4] = {};

  async_g2l_b128(bdst0, bsrc);
  async_g2l_b128(bdst0 + 16, (const char*)bsrc + 16);

  for (int k0 = 0; k0 < K; k0 += 32) {
    const int cur = (k0 >> 5) & 1;
    if (k0 + 32 < K) {
      const unsigned nd = cur ? bdst0 : bdst1;
      const _Float16* ns = bsrc + (k0 + 32);
      async_g2l_b128(nd, ns);
      async_g2l_b128(nd + 16, (const char*)ns + 16);
      asm volatile("s_wait_asynccnt 2" ::: "memory");
    } else {
      asm volatile("s_wait_asynccnt 0" ::: "memory");
    }
    __syncthreads();

    if (k0 + 64 < K) __builtin_prefetch(arow + k0 + 64, 0, 1);

    const int d0 = k0 + 16 * hi;
    const h8 a = *(const h8*)(arow + d0);
    const h8 c = *(const h8*)(arow + d0 + 8);
    const v16h af = pack_afrag(a, c);
#pragma unroll
    for (int t = 0; t < 4; ++t) {
      const v16h bf = *(const v16h*)(&bstage[cur][(16 * t + lo) * 32 + 16 * hi]);
      acc[t] = WMMA16(af, bf, acc[t]);
    }
    __syncthreads();
  }

#pragma unroll
  for (int t = 0; t < 4; ++t) {
    const int n = N0 + 16 * t + lo;
    const float bv = bias[n];
#pragma unroll
    for (int r = 0; r < 8; ++r) {
      const int row = M0 + r + 8 * hi;
      out[(size_t)row * 1024 + n] = acc[t][r] + bv;
    }
  }
}

// ------------------------------ launcher -----------------------------------
extern "C" void kernel_launch(void* const* d_in, const int* in_sizes, int n_in,
                              void* d_out, int out_size, void* d_ws, size_t ws_size,
                              hipStream_t stream) {
  (void)in_sizes; (void)n_in; (void)out_size; (void)ws_size;
  const float* hidden   = (const float*)d_in[0];   // [2,2048,1024]
  const float* c_attn_w = (const float*)d_in[1];   // [1024,3072]
  const float* c_attn_b = (const float*)d_in[2];   // [3072]
  const float* c_proj_w = (const float*)d_in[3];   // [1024,1024]
  const float* c_proj_b = (const float*)d_in[4];   // [1024]
  float* out = (float*)d_out;                      // [2,2048,1024] f32

  // Workspace carve-up (f16 elements): ~40 MB total.
  _Float16* ws      = (_Float16*)d_ws;
  _Float16* wqkv_t  = ws;                             // [3072][1024]
  _Float16* wproj_t = wqkv_t + (size_t)3072 * 1024;   // [1024][1024]
  _Float16* qws     = wproj_t + (size_t)1024 * 1024;  // [2,16,2048,64]
  _Float16* kws     = qws + (size_t)4194304;          // [2,16,2048,64]
  _Float16* vws     = kws + (size_t)4194304;          // [2,16,64,2048] (transposed)
  _Float16* ows     = vws + (size_t)4194304;          // [2,2048,1024]

  transpose_cvt<<<dim3(3072 / 32, 1024 / 32), 256, 0, stream>>>(c_attn_w, wqkv_t, 1024, 3072);
  transpose_cvt<<<dim3(1024 / 32, 1024 / 32), 256, 0, stream>>>(c_proj_w, wproj_t, 1024, 1024);
  qkv_gemm<<<dim3(4096 / 64, 3072 / 64), 128, 0, stream>>>(hidden, wqkv_t, c_attn_b,
                                                           qws, kws, vws);
  attn_kernel<<<1024, 128, 0, stream>>>(qws, kws, vws, ows);
  proj_gemm<<<dim3(4096 / 64, 1024 / 64), 128, 0, stream>>>(ows, wproj_t, c_proj_b, out);
}
